// scGraph_26585847562902
// MI455X (gfx1250) — matrix operand order repaired and checked by reading, not compiled, not run on hardware
//
#include <hip/hip_runtime.h>

// ---------------------------------------------------------------- constants
#define NN    2207          // nodes
#define BB    512           // batch
#define EE    151215        // edges
#define MIDC  8
#define GC1C  12
#define GC2C  4
#define FC1C  256
#define FC2C  64
#define KFC1  8828          // 4*NN
#define KPAD  8832          // KFC1 padded to multiple of 32
#define EPS_F 1e-5f

typedef __attribute__((ext_vector_type(16))) _Float16 v16h;
typedef __attribute__((ext_vector_type(8)))  _Float16 v8h;
typedef __attribute__((ext_vector_type(8)))  float    v8f;

// ---------------------------------------------------------------- stage A: edge scatter
__global__ __launch_bounds__(256) void edge_scatter_kernel(
    const float* __restrict__ x, const int* __restrict__ ei,
    const float* __restrict__ ewp, float* __restrict__ agg, float* __restrict__ deg)
{
    int e   = blockIdx.x;
    int src = ei[e];
    int dst = ei[EE + e];
    float ew = 1.0f / (1.0f + __expf(-ewp[e]));
    const float* xs = x   + (size_t)src * BB;
    float*       ad = agg + (size_t)dst * BB;
    int b = threadIdx.x;
    atomicAdd(&ad[b],        xs[b]        * ew);
    atomicAdd(&ad[b + 256],  xs[b + 256]  * ew);
    if (threadIdx.x == 0) atomicAdd(&deg[dst], 1.0f);
}

// ---------------------------------------------------------------- stage B: mean-agg + SAGE + relu
__global__ __launch_bounds__(256) void sage_kernel(
    const float* __restrict__ agg, const float* __restrict__ deg,
    const float* __restrict__ sw, const float* __restrict__ sb,
    float* __restrict__ h)  // h layout: [B][N][8]
{
    int i = blockIdx.x * 256 + threadIdx.x;     // i = n*BB + b  (N*B exact multiple of 256)
    int n = i >> 9;
    int b = i & (BB - 1);
    float a = agg[i] / fmaxf(deg[n], 1.0f);
    float* hp = h + ((size_t)b * NN + n) * MIDC;
#pragma unroll
    for (int c = 0; c < MIDC; ++c)
        hp[c] = fmaxf(a * sw[c] + sb[c], 0.0f);
}

// ---------------------------------------------------------------- stage C: per-batch LN stats
__global__ __launch_bounds__(256) void ln_stats_kernel(
    const float* __restrict__ h, float* __restrict__ mu, float* __restrict__ rstd)
{
    int b = blockIdx.x;
    const float* p = h + (size_t)b * NN * MIDC;
    float s = 0.f, ss = 0.f;
    for (int j = threadIdx.x; j < NN * MIDC; j += 256) {
        float v = p[j]; s += v; ss += v * v;
    }
    __shared__ float sh1[256], sh2[256];
    sh1[threadIdx.x] = s; sh2[threadIdx.x] = ss;
    __syncthreads();
    for (int o = 128; o > 0; o >>= 1) {
        if (threadIdx.x < o) { sh1[threadIdx.x] += sh1[threadIdx.x + o];
                               sh2[threadIdx.x] += sh2[threadIdx.x + o]; }
        __syncthreads();
    }
    if (threadIdx.x == 0) {
        float inv = 1.0f / (float)(NN * MIDC);
        float m = sh1[0] * inv;
        float v = sh2[0] * inv - m * m;
        mu[b]   = m;
        rstd[b] = rsqrtf(fmaxf(v, 0.f) + EPS_F);
    }
}

// ---------------------------------------------------------------- stage D: LN apply + gc1/bn1 + gc2/bn2 -> f16 flat
__global__ __launch_bounds__(256) void conv_fused_kernel(
    const float* __restrict__ h, const float* __restrict__ mu, const float* __restrict__ rstd,
    const float* __restrict__ lng, const float* __restrict__ lnb,
    const float* __restrict__ gc1w, const float* __restrict__ gc1b,
    const float* __restrict__ bn1g, const float* __restrict__ bn1b,
    const float* __restrict__ gc2w, const float* __restrict__ gc2b,
    const float* __restrict__ bn2g, const float* __restrict__ bn2b,
    _Float16* __restrict__ flat)   // [B][KPAD], channel-major p*NN+n
{
    __shared__ float sW1[GC1C * MIDC], sB1[GC1C], sS1[GC1C], sO1[GC1C];
    __shared__ float sW2[GC2C * GC1C], sB2[GC2C], sS2[GC2C], sO2[GC2C];
    int t = threadIdx.x;
    const float rs = rsqrtf(1.0f + EPS_F);
    if (t < GC1C * MIDC) sW1[t] = gc1w[t];
    if (t < GC2C * GC1C) sW2[t] = gc2w[t];
    if (t < GC1C) { sB1[t] = gc1b[t]; sS1[t] = bn1g[t] * rs; sO1[t] = bn1b[t]; }
    if (t < GC2C) { sB2[t] = gc2b[t]; sS2[t] = bn2g[t] * rs; sO2[t] = bn2b[t]; }
    __syncthreads();

    int n = blockIdx.x * 256 + t;
    int b = blockIdx.y;
    if (n >= NN) return;
    float m = mu[b], r = rstd[b];
    const float* hp = h   + ((size_t)b * NN + n) * MIDC;
    const float* lg = lng + (size_t)n * MIDC;
    const float* lo = lnb + (size_t)n * MIDC;

    float g[MIDC];
#pragma unroll
    for (int c = 0; c < MIDC; ++c)
        g[c] = (hp[c] - m) * r * lg[c] + lo[c];

    float y1[GC1C];
#pragma unroll
    for (int o = 0; o < GC1C; ++o) {
        float acc = sB1[o];
#pragma unroll
        for (int c = 0; c < MIDC; ++c) acc += sW1[o * MIDC + c] * g[c];
        y1[o] = fmaxf(acc, 0.0f) * sS1[o] + sO1[o];   // relu then bn (matches reference)
    }
#pragma unroll
    for (int p = 0; p < GC2C; ++p) {
        float acc = sB2[p];
#pragma unroll
        for (int o = 0; o < GC1C; ++o) acc += sW2[p * GC1C + o] * y1[o];
        float v = fmaxf(acc, 0.0f) * sS2[p] + sO2[p];
        flat[(size_t)b * KPAD + p * NN + n] = (_Float16)v;
    }
}

// ---------------------------------------------------------------- f32 weights -> f16 in WMMA B-fragment order
// src: [srcK x N] f32 row-major. dst: chunk c = ((nT*(Kpad/32) + kt)*32 + lane)
// holds 16 contiguous halfs = B[kt*32 + 8*hi + (j<8 ? j : j+8)][nT*16 + lm], j = 0..15
// (lane = hi*16 + lm). In the GEMM each lane then loads its fragment as one
// contiguous, 32B-aligned 16-half chunk -> two global_load_b128.
__global__ __launch_bounds__(256) void cvt_swizzle_b_kernel(
    const float* __restrict__ src, _Float16* __restrict__ dst,
    int srcK, int Kpad, int N)
{
    int nK = Kpad >> 5;
    int total = (N >> 4) * nK * 32;
    int c = blockIdx.x * 256 + threadIdx.x;
    if (c >= total) return;
    int lane = c & 31;
    int kt   = (c >> 5) % nK;
    int nT   = c / (nK << 5);
    int lm = lane & 15, hi = lane >> 4;
    int n     = nT * 16 + lm;
    int kbase = kt * 32 + 8 * hi;
    v16h frag;
#pragma unroll
    for (int j = 0; j < 16; ++j) {
        int k = kbase + (j < 8 ? j : j + 8);
        frag[j] = (k < srcK) ? (_Float16)src[(size_t)k * N + n] : (_Float16)0.0f;
    }
    *((v16h*)dst + c) = frag;
}

// ---------------------------------------------------------------- WMMA GEMM: out = relu(bn(A@B + bias))
// A:   [M x K] f16 row-major (lda = K), 16B-aligned rows
// Bsw: swizzled fragments from cvt_swizzle_b_kernel
// one 16x16 C tile per wave, 8 waves/block; grid = (N/16, M/128)
// Fragment layouts per CDNA5 ISA 7.12.2 (wave32):
//   A 16x32 f16 : lanes 0-15 -> K {0..7,16..23}, lanes 16-31 -> K {8..15,24..31}
//   B 32x16 f16 : lanes 0-15 -> K {0..7,16..23}, lanes 16-31 -> K {8..15,24..31}
//   D 16x16 f32 : VGPR r -> M = r + 8*hi, N = lane&15
template <bool HALF_OUT>
__global__ __launch_bounds__(256) void wmma_fc_kernel(
    const _Float16* __restrict__ A, const _Float16* __restrict__ Bsw,
    const float* __restrict__ bias, const float* __restrict__ gamma,
    const float* __restrict__ beta,
    _Float16* __restrict__ outH, float* __restrict__ outF,
    int K, int ldn)
{
    int lane = threadIdx.x & 31;
    int w    = threadIdx.x >> 5;
    int lm   = lane & 15;
    int hi   = lane >> 4;
    int mTile = (blockIdx.y * 8 + w) * 16;
    int nBase = blockIdx.x * 16;
    int nK    = K >> 5;

    const _Float16* ap = A + (size_t)(mTile + lm) * K + 8 * hi;       // 16B aligned
    const _Float16* bp = Bsw + ((size_t)blockIdx.x * nK * 32 + lane) * 16; // 32B aligned
    v8f acc = {0.f, 0.f, 0.f, 0.f, 0.f, 0.f, 0.f, 0.f};

#pragma unroll 2
    for (int kt = 0; kt < nK; ++kt) {
        v8h a0 = *(const v8h*)(ap);        // K = k0 + 8*hi + 0..7
        v8h a1 = *(const v8h*)(ap + 16);   // K = k0 + 16 + 8*hi + 0..7
        v16h bfrag = *(const v16h*)(bp);   // K = k0 + 8*hi + {0..7,16..23} at column nBase+lm
        v16h afrag;
#pragma unroll
        for (int j = 0; j < 8; ++j) { afrag[j] = a0[j]; afrag[8 + j] = a1[j]; }

        acc = __builtin_amdgcn_wmma_f32_16x16x32_f16(
            /*neg_a=*/false, afrag, /*neg_b=*/false, bfrag,
            /*c_mod=*/(short)0, acc, /*reuse_a=*/false, /*reuse_b=*/false);

        ap += 32;          // next 32 K values
        bp += 32 * 16;     // next k-step's 32 fragments
    }

    int n = nBase + lm;
    float scale = gamma[n] * rsqrtf(1.0f + EPS_F);
    float shift = beta[n];
    float bs    = bias[n];
#pragma unroll
    for (int r = 0; r < 8; ++r) {
        int m = mTile + r + 8 * hi;
        float v = acc[r] + bs;
        v = v * scale + shift;        // FC path: bias -> bn -> relu (matches reference)
        v = fmaxf(v, 0.0f);
        if (HALF_OUT) outH[(size_t)m * ldn + n] = (_Float16)v;
        else          outF[(size_t)m * ldn + n] = v;
    }
}

// ---------------------------------------------------------------- head: [512x64]@[64x2] + softmax
__global__ __launch_bounds__(256) void head_kernel(
    const float* __restrict__ z2, const float* __restrict__ ow,
    const float* __restrict__ ob, float* __restrict__ out)
{
    int m = blockIdx.x * 256 + threadIdx.x;
    if (m >= BB) return;
    const float* z = z2 + (size_t)m * FC2C;
    float l0 = ob[0], l1 = ob[1];
#pragma unroll 8
    for (int k = 0; k < FC2C; ++k) {
        float v = z[k];
        l0 += v * ow[2 * k];
        l1 += v * ow[2 * k + 1];
    }
    float mx = fmaxf(l0, l1);
    float e0 = __expf(l0 - mx), e1 = __expf(l1 - mx);
    float inv = 1.0f / (e0 + e1);
    out[2 * m]     = e0 * inv;
    out[2 * m + 1] = e1 * inv;
}

// ---------------------------------------------------------------- launch
extern "C" void kernel_launch(void* const* d_in, const int* in_sizes, int n_in,
                              void* d_out, int out_size, void* d_ws, size_t ws_size,
                              hipStream_t stream)
{
    const float* x      = (const float*)d_in[0];
    const int*   ei     = (const int*)  d_in[1];
    const float* ewp    = (const float*)d_in[2];
    const float* sw     = (const float*)d_in[3];
    // d_in[4] = sage_self_weight (multiplied by 0 in reference -> unused)
    const float* sb     = (const float*)d_in[5];
    const float* lng    = (const float*)d_in[6];
    const float* lnb    = (const float*)d_in[7];
    const float* gc1w   = (const float*)d_in[8];
    const float* gc1b   = (const float*)d_in[9];
    const float* bn1g   = (const float*)d_in[10];
    const float* bn1b   = (const float*)d_in[11];
    const float* gc2w   = (const float*)d_in[12];
    const float* gc2b   = (const float*)d_in[13];
    const float* bn2g   = (const float*)d_in[14];
    const float* bn2b   = (const float*)d_in[15];
    const float* fc1w   = (const float*)d_in[16];
    const float* fc1b   = (const float*)d_in[17];
    const float* fbn1g  = (const float*)d_in[18];
    const float* fbn1b  = (const float*)d_in[19];
    const float* fc2w   = (const float*)d_in[20];
    const float* fc2b   = (const float*)d_in[21];
    const float* fbn2g  = (const float*)d_in[22];
    const float* fbn2b  = (const float*)d_in[23];
    const float* outw   = (const float*)d_in[24];
    const float* outb   = (const float*)d_in[25];
    float* out = (float*)d_out;

    // ---- workspace carve (256B aligned)
    char* ws = (char*)d_ws;
    size_t off = 0;
    auto carve = [&](size_t bytes) -> void* {
        void* p = ws + off;
        off += (bytes + 255) & ~(size_t)255;
        return p;
    };
    float*     agg    = (float*)    carve((size_t)NN * BB * 4);          // 4.5 MB
    float*     deg    = (float*)    carve((size_t)NN * 4);
    float*     h      = (float*)    carve((size_t)BB * NN * MIDC * 4);   // 36 MB
    float*     mu     = (float*)    carve((size_t)BB * 4);
    float*     rstd   = (float*)    carve((size_t)BB * 4);
    _Float16*  flatH  = (_Float16*) carve((size_t)BB * KPAD * 2);        // 9 MB
    _Float16*  fc1wS  = (_Float16*) carve((size_t)KPAD * FC1C * 2);      // 4.5 MB (swizzled)
    _Float16*  z1H    = (_Float16*) carve((size_t)BB * FC1C * 2);
    _Float16*  fc2wS  = (_Float16*) carve((size_t)FC1C * FC2C * 2);      // swizzled
    float*     z2     = (float*)    carve((size_t)BB * FC2C * 4);
    (void)ws_size; (void)in_sizes; (void)n_in; (void)out_size;

    // ---- zero-init accumulators + padded f16 activation buffer
    hipMemsetAsync(agg,   0, (size_t)NN * BB * 4, stream);
    hipMemsetAsync(deg,   0, (size_t)NN * 4, stream);
    hipMemsetAsync(flatH, 0, (size_t)BB * KPAD * 2, stream);

    // ---- weight conversion to f16, pre-swizzled into WMMA B-fragment order
    {
        int chunks1 = (FC1C / 16) * (KPAD / 32) * 32;    // fc1: N=256, Kpad=8832
        cvt_swizzle_b_kernel<<<dim3((chunks1 + 255) / 256), 256, 0, stream>>>(
            fc1w, fc1wS, KFC1, KPAD, FC1C);
        int chunks2 = (FC2C / 16) * (FC1C / 32) * 32;    // fc2: N=64, K=256
        cvt_swizzle_b_kernel<<<dim3((chunks2 + 255) / 256), 256, 0, stream>>>(
            fc2w, fc2wS, FC1C, FC1C, FC2C);
    }

    // ---- A: edge gather-weight-scatter (atomic mean-agg)
    edge_scatter_kernel<<<dim3(EE), 256, 0, stream>>>(x, ei, ewp, agg, deg);

    // ---- B: normalize + SAGE + relu
    sage_kernel<<<dim3((NN * BB) / 256), 256, 0, stream>>>(agg, deg, sw, sb, h);

    // ---- C: per-batch LayerNorm statistics
    ln_stats_kernel<<<dim3(BB), 256, 0, stream>>>(h, mu, rstd);

    // ---- D: LN apply + gc1/bn1 + gc2/bn2, emit f16 flat [B, KPAD]
    conv_fused_kernel<<<dim3((NN + 255) / 256, BB), 256, 0, stream>>>(
        h, mu, rstd, lng, lnb, gc1w, gc1b, bn1g, bn1b, gc2w, gc2b, bn2g, bn2b, flatH);

    // ---- E: FC1 via WMMA  [512 x 8832] @ [8832 x 256] -> f16 [512 x 256]
    wmma_fc_kernel<true><<<dim3(FC1C / 16, BB / 128), 256, 0, stream>>>(
        flatH, fc1wS, fc1b, fbn1g, fbn1b, z1H, nullptr, KPAD, FC1C);

    // ---- F: FC2 via WMMA  [512 x 256] @ [256 x 64] -> f32 [512 x 64]
    wmma_fc_kernel<false><<<dim3(FC2C / 16, BB / 128), 256, 0, stream>>>(
        z1H, fc2wS, fc2b, fbn2g, fbn2b, nullptr, z2, FC1C, FC2C);

    // ---- G: output projection + softmax
    head_kernel<<<dim3(BB / 256), 256, 0, stream>>>(z2, outw, outb, out);
}
